// Model_9612136808826
// MI455X (gfx1250) — matrix operand order repaired and checked by reading, not compiled, not run on hardware
//
#include <hip/hip_runtime.h>
#include <hip/hip_bf16.h>
#include <math.h>

typedef __bf16 bf16;
typedef __attribute__((ext_vector_type(16))) __bf16 v16bf;
typedef __attribute__((ext_vector_type(8)))  float  v8f;

#define HID 256
#define BATCH 8
#define TSTEPS 256
#define VOCAB 50257

// ---------------------------------------------------------------------------
// WMMA fragment loaders (bf16, 16x16x32 shape), per CDNA5 ISA VGPR layouts.
// A (16x32, MxK): lane m = lane%16; lanes 0-15 hold K {0..7,16..23},
//                 lanes 16-31 hold K {8..15,24..31}. Two 16B loads.
// B (32x16, KxN): lane n = lane%16; lanes 0-15 hold K {0..7,16..23},
//                 lanes 16-31 hold K {8..15,24..31}. Two 16B loads (row n of W).
// ---------------------------------------------------------------------------
__device__ inline v16bf load_frag_a(const bf16* __restrict__ p, int ld,
                                    int m0, int k0, int lane) {
  int m  = m0 + (lane & 15);
  int kb = k0 + ((lane >> 4) << 3);          // +0 or +8
  const bf16* q = p + (size_t)m * ld + kb;
  union { v16bf v; float4 f[2]; } u;
  u.f[0] = *(const float4*)(q);              // K offs +0..7
  u.f[1] = *(const float4*)(q + 16);         // K offs +16..23
  return u.v;
}

__device__ inline v16bf load_frag_b(const bf16* __restrict__ p, int ld,
                                    int n0, int k0, int lane, int nmax) {
  int n = n0 + (lane & 15);
  if (n >= nmax) n = nmax - 1;               // tail clamp (result masked at store)
  int kb = k0 + ((lane >> 4) << 3);          // +0 or +8
  const bf16* q = p + (size_t)n * ld + kb;
  union { v16bf v; float4 f[2]; } u;
  u.f[0] = *(const float4*)(q);
  u.f[1] = *(const float4*)(q + 16);
  return u.v;
}

__device__ inline v8f wmma_bf16(v16bf a, v16bf b, v8f c) {
  return __builtin_amdgcn_wmma_f32_16x16x32_bf16(false, a, false, b,
                                                 (short)0, c, false, false);
}

// ---------------------------------------------------------------------------
// Embedding gather -> bf16 rows [B*T, HID]
// ---------------------------------------------------------------------------
__global__ void k_gather_embed(const int* __restrict__ x,
                               const float* __restrict__ emb,
                               bf16* __restrict__ xe) {
  int m = blockIdx.x;                         // m = b*T + t (x is [B,T] row-major)
  int k = threadIdx.x;
  int tok = x[m];
  xe[(size_t)m * HID + k] = (bf16)emb[(size_t)tok * HID + k];
}

__global__ void k_f32_to_bf16(const float* __restrict__ s,
                              bf16* __restrict__ d, int n) {
  int i = blockIdx.x * blockDim.x + threadIdx.x;
  if (i < n) d[i] = (bf16)s[i];
}

// ---------------------------------------------------------------------------
// C[M,N] = A[M,K=HID] @ Bm[N,K=HID]^T + bias0 (+ bias1)
// Block = 256 threads = 8 waves. Each wave owns a 64x64 tile (4x4 WMMA tiles):
// per K-step, 4 A-frags + 4 B-frags (16x16B loads) feed 16 WMMAs -> 1.0
// loads/WMMA, 4x register reuse. Block covers 64M x 512N; grid.x steps N by
// 512, grid.y steps M by 64. N tail (50257) clamped on load, guarded on store.
// ---------------------------------------------------------------------------
__global__ __launch_bounds__(256) void k_gemm_bf16(
    const bf16* __restrict__ A, const bf16* __restrict__ Bm,
    const float* __restrict__ bias0, const float* __restrict__ bias1,
    float* __restrict__ C, int M, int N) {
  const int lane = threadIdx.x & 31;
  const int wave = threadIdx.x >> 5;
  const int m0 = blockIdx.y * 64;
  const int n0 = blockIdx.x * 512 + wave * 64;
  const int cl = lane & 15;
  const int rbase = (lane >> 4) << 3;          // 0 or 8

  v8f acc[4][4];
  for (int j = 0; j < 4; ++j) {
    int col = n0 + j * 16 + cl;
    float bv = 0.f;
    if (col < N) { bv = bias0[col]; if (bias1) bv += bias1[col]; }
    for (int i = 0; i < 4; ++i)
      for (int r = 0; r < 8; ++r) acc[i][j][r] = bv;
  }

  for (int k0 = 0; k0 < HID; k0 += 32) {
    v16bf a[4];
#pragma unroll
    for (int i = 0; i < 4; ++i)
      a[i] = load_frag_a(A, HID, m0 + 16 * i, k0, lane);
#pragma unroll
    for (int j = 0; j < 4; ++j) {
      v16bf b = load_frag_b(Bm, HID, n0 + 16 * j, k0, lane, N);
#pragma unroll
      for (int i = 0; i < 4; ++i)
        acc[i][j] = wmma_bf16(a[i], b, acc[i][j]);
    }
  }

  for (int i = 0; i < 4; ++i)
    for (int j = 0; j < 4; ++j) {
      int col = n0 + j * 16 + cl;
      if (col < N) {
        int row0 = m0 + i * 16 + rbase;
        for (int r = 0; r < 8; ++r)
          C[(size_t)(row0 + r) * N + col] = acc[i][j][r];
      }
    }
}

// ---------------------------------------------------------------------------
// LSTM scan: single workgroup, 512 threads = 16 waves.
// gates[8,1024] = xg_t + h @ W_hh^T via WMMA (batch 8 padded to M=16).
// h (bf16), c (f32), gates (f32) live in LDS; 2 barriers per timestep.
// ---------------------------------------------------------------------------
__global__ __launch_bounds__(512) void k_lstm_scan(
    const float* __restrict__ xg,     // [B*T, 4H]
    const bf16* __restrict__ Whh,     // [4H, H] bf16
    bf16* __restrict__ hs,            // [B*T, H] bf16 out
    float* __restrict__ hN, float* __restrict__ cN) {
  __shared__ __align__(16) bf16 h_sh[16 * HID];       // rows 8..15 stay zero
  __shared__ float c_sh[BATCH * HID];
  __shared__ float g_sh[BATCH * 4 * HID];

  const int tid  = threadIdx.x;
  const int lane = tid & 31;
  const int wave = tid >> 5;
  const int cl    = lane & 15;
  const int rbase = (lane >> 4) << 3;

  for (int i = tid; i < 16 * HID; i += 512) h_sh[i] = (bf16)0.f;
  for (int i = tid; i < BATCH * HID; i += 512) c_sh[i] = 0.f;
  __syncthreads();

  for (int t = 0; t < TSTEPS; ++t) {
    // A fragments of h (shared across the wave's 4 N-tiles)
    v16bf afr[8];
#pragma unroll
    for (int kk = 0; kk < 8; ++kk)
      afr[kk] = load_frag_a(h_sh, HID, 0, kk * 32, lane);

    for (int nt = 0; nt < 4; ++nt) {
      const int n0 = wave * 64 + nt * 16;
      const int col = n0 + cl;
      v8f acc;
      for (int r = 0; r < 8; ++r) {
        int b = rbase + r;
        int bb = b < BATCH ? b : (BATCH - 1);           // rows 8..15: dont-care
        acc[r] = xg[((size_t)(bb * TSTEPS + t)) * (4 * HID) + col];
      }
#pragma unroll
      for (int kk = 0; kk < 8; ++kk) {
        v16bf bfr = load_frag_b(Whh, HID, n0, kk * 32, lane, 4 * HID);
        acc = wmma_bf16(afr[kk], bfr, acc);
      }
      if (lane < 16) {                                   // rows 0..7 == batch
        for (int r = 0; r < 8; ++r)
          g_sh[r * (4 * HID) + col] = acc[r];
      }
    }
    __syncthreads();

    // elementwise gate math: 2048 cells / 512 threads
    for (int j = 0; j < 4; ++j) {
      int idx = tid + j * 512;
      int b = idx >> 8, k = idx & 255;
      float gi = g_sh[b * (4 * HID) +            k];
      float gf = g_sh[b * (4 * HID) +  HID     + k];
      float gg = g_sh[b * (4 * HID) + 2 * HID  + k];
      float go = g_sh[b * (4 * HID) + 3 * HID  + k];
      float si = 1.f / (1.f + __expf(-gi));
      float sf = 1.f / (1.f + __expf(-gf));
      float so = 1.f / (1.f + __expf(-go));
      float cv = sf * c_sh[b * HID + k] + si * tanhf(gg);
      float hv = so * tanhf(cv);
      c_sh[b * HID + k] = cv;
      h_sh[b * HID + k] = (bf16)hv;
      hs[((size_t)(b * TSTEPS + t)) * HID + k] = (bf16)hv;
      if (t == TSTEPS - 1) {
        hN[b * HID + k] = hv;
        cN[b * HID + k] = cv;
      }
    }
    __syncthreads();
  }
}

// ---------------------------------------------------------------------------
extern "C" void kernel_launch(void* const* d_in, const int* in_sizes, int n_in,
                              void* d_out, int out_size, void* d_ws, size_t ws_size,
                              hipStream_t stream) {
  const int*   x    = (const int*)  d_in[0];
  const float* emb  = (const float*)d_in[1];
  const float* W_ih = (const float*)d_in[2];
  const float* W_hh = (const float*)d_in[3];
  const float* b_ih = (const float*)d_in[4];
  const float* b_hh = (const float*)d_in[5];
  const float* W_fc = (const float*)d_in[6];
  const float* b_fc = (const float*)d_in[7];

  float* out = (float*)d_out;                        // [2048, VOCAB]
  float* hN  = out + (size_t)BATCH * TSTEPS * VOCAB; // [1,8,256]
  float* cN  = hN + BATCH * HID;                     // [1,8,256]

  // workspace carve-up (all offsets 512B-aligned)
  char* ws = (char*)d_ws;
  bf16* xe_bf  = (bf16*)ws;  ws += (size_t)BATCH * TSTEPS * HID * 2;  // 1 MB
  bf16* Wih_bf = (bf16*)ws;  ws += (size_t)4 * HID * HID * 2;         // 512 KB
  bf16* Whh_bf = (bf16*)ws;  ws += (size_t)4 * HID * HID * 2;         // 512 KB
  bf16* Wfc_bf = (bf16*)ws;  ws += (size_t)VOCAB * HID * 2;           // 25.7 MB
  float* xg    = (float*)ws; ws += (size_t)BATCH * TSTEPS * 4 * HID * 4; // 8 MB
  bf16* hs_bf  = (bf16*)ws;                                           // 1 MB

  const int M = BATCH * TSTEPS;                      // 2048

  // 1) embedding gather -> bf16
  k_gather_embed<<<M, HID, 0, stream>>>(x, emb, xe_bf);

  // 2) weight conversions f32 -> bf16
  k_f32_to_bf16<<<(4 * HID * HID + 255) / 256, 256, 0, stream>>>(W_ih, Wih_bf, 4 * HID * HID);
  k_f32_to_bf16<<<(4 * HID * HID + 255) / 256, 256, 0, stream>>>(W_hh, Whh_bf, 4 * HID * HID);
  k_f32_to_bf16<<<(VOCAB * HID + 255) / 256, 256, 0, stream>>>(W_fc, Wfc_bf, VOCAB * HID);

  // 3) input projection: xg[M,4H] = xe @ W_ih^T + (b_ih + b_hh)
  //    block = 64M x 512N -> grid (1024/512, 2048/64)
  k_gemm_bf16<<<dim3((4 * HID) / 512, M / 64), 256, 0, stream>>>(
      xe_bf, Wih_bf, b_ih, b_hh, xg, M, 4 * HID);

  // 4) sequential LSTM scan (single WGP-resident workgroup)
  k_lstm_scan<<<1, 512, 0, stream>>>(xg, Whh_bf, hs_bf, hN, cN);

  // 5) output projection: out[M,VOCAB] = hs @ W_fc^T + b_fc
  k_gemm_bf16<<<dim3((VOCAB + 511) / 512, M / 64), 256, 0, stream>>>(
      hs_bf, Wfc_bf, b_fc, nullptr, out, M, VOCAB);
}
